// LstmModel_47674136986214
// MI455X (gfx1250) — compile-verified, matrix operand order checked by reading, not run on hardware
//
#include <hip/hip_runtime.h>

// ---------------------------------------------------------------------------
// Types for WMMA bf16 (gfx1250): A/B = 16 bf16 per lane (16x32 tile),
// C/D = 8 f32 per lane (16x16 tile).
// ---------------------------------------------------------------------------
typedef __bf16 bf16;
typedef __attribute__((ext_vector_type(16))) __bf16 v16bf;
typedef __attribute__((ext_vector_type(8)))  __bf16 v8bf;
typedef __attribute__((ext_vector_type(8)))  float   v8f;

// ---------------------------------------------------------------------------
// Fragment loader: a 16x32 bf16 tile from row-major memory with row stride
// `ld` (elements).  Per ISA layout: lane L -> row (L&15); holds K values
// kb..kb+7 and kb+16..kb+23 where kb = (L>>4)*8.  Both halves are contiguous
// 16-byte chunks -> two b128 loads per lane.  The same layout serves the
// B operand when the weight is stored row-major as W[n][k] (lane = column n).
// ---------------------------------------------------------------------------
__device__ __forceinline__ v16bf load_tile_frag(const bf16* __restrict__ base, int ld) {
  const int lane = threadIdx.x & 31;
  const int r  = lane & 15;
  const int kb = (lane >> 4) << 3;           // 0 or 8
  const bf16* p = base + (size_t)r * ld + kb;
  v8bf lo = *(const v8bf*)(p);
  v8bf hi = *(const v8bf*)(p + 16);
  v16bf f;
#pragma unroll
  for (int i = 0; i < 8; ++i) { f[i] = lo[i]; f[i + 8] = hi[i]; }
  return f;
}

__device__ __forceinline__ float sigmoidf_(float x) {
  return 1.0f / (1.0f + __expf(-x));
}

// ---------------------------------------------------------------------------
// f32 -> bf16 conversion
// ---------------------------------------------------------------------------
__global__ void f2bf_kernel(const float* __restrict__ in, bf16* __restrict__ out, int n) {
  int i = blockIdx.x * blockDim.x + threadIdx.x;
  if (i < n) out[i] = (bf16)in[i];
}

// ---------------------------------------------------------------------------
// Generic bf16 WMMA GEMM:  Out[M x N] = A[M x K] * W[N x K]^T   (f32 accum)
// K is compile-time -> unrolled k-loop, loads use immediate offsets off fixed
// per-lane base pointers (no per-iter 64-bit address VALU), B fragments of a
// k-step preloaded into distinct registers so WMMAs overlap loads.
// Register macro-tile: each wave owns 2 M-tiles x 4 N-tiles (8 WMMAs per
// k-step from 6 fragment loads -> ~22 FLOP/byte of cache traffic).
// Block = 8 waves -> covers 32 (M) x 512 (N).  Requires M%32==0, N%512==0.
//   EPI = 0 : store raw f32 (gate pre-activations for recurrent layers)
//   EPI = 1 : += bias, ReLU, store bf16 (MLP layers)
// ---------------------------------------------------------------------------
template <int EPI, int K>
__global__ __launch_bounds__(256) void gemm_bf16_wmma(
    const bf16* __restrict__ A, const bf16* __restrict__ W,
    float* __restrict__ outF, bf16* __restrict__ outB,
    const float* __restrict__ bias, int M, int N)
{
  const int wave = threadIdx.x >> 5;
  const int lane = threadIdx.x & 31;
  const int m0 = blockIdx.y << 5;                    // 2 M-tiles per wave
  const int n0 = (blockIdx.x << 9) + (wave << 6);    // 4 N-tiles per wave

  v8f acc[2][4];
#pragma unroll
  for (int mi = 0; mi < 2; ++mi)
#pragma unroll
    for (int j = 0; j < 4; ++j) {
      v8f z = {0.f, 0.f, 0.f, 0.f, 0.f, 0.f, 0.f, 0.f};
      acc[mi][j] = z;
    }

  const bf16* A0 = A + (size_t)m0 * K;
  const bf16* A1 = A + (size_t)(m0 + 16) * K;
  const bf16* W0 = W + (size_t)n0 * K;
#pragma unroll
  for (int k0 = 0; k0 < K; k0 += 32) {
    v16bf a0 = load_tile_frag(A0 + k0, K);
    v16bf a1 = load_tile_frag(A1 + k0, K);
    v16bf b[4];
#pragma unroll
    for (int j = 0; j < 4; ++j)
      b[j] = load_tile_frag(W0 + (size_t)j * 16 * K + k0, K);
#pragma unroll
    for (int j = 0; j < 4; ++j) {
      acc[0][j] = __builtin_amdgcn_wmma_f32_16x16x32_bf16(
          false, a0, false, b[j], (short)0, acc[0][j], false, false);
      acc[1][j] = __builtin_amdgcn_wmma_f32_16x16x32_bf16(
          false, a1, false, b[j], (short)0, acc[1][j], false, false);
    }
  }

  // C/D layout: lane -> col (lane&15); VGPR e -> row e + 8*(lane>>4)
  const int cn = lane & 15;
  const int rb = (lane >> 4) << 3;
#pragma unroll
  for (int mi = 0; mi < 2; ++mi) {
#pragma unroll
    for (int j = 0; j < 4; ++j) {
      const int col = n0 + j * 16 + cn;
#pragma unroll
      for (int e = 0; e < 8; ++e) {
        const int row = m0 + mi * 16 + rb + e;
        float v = acc[mi][j][e];
        if (EPI == 0) {
          outF[(size_t)row * N + col] = v;
        } else {
          v += bias[col];
          v = v > 0.f ? v : 0.f;
          outB[(size_t)row * N + col] = (bf16)v;
        }
      }
    }
  }
}

// ---------------------------------------------------------------------------
// Fully fused non-recurrent "lstm1" layer (f gate is dead in the reference,
// so it is skipped entirely -> 25% fewer FLOPs and no f32 gate round-trip):
//   h = sigmoid(o+b) * tanh( sigmoid(i+b) * tanh(g+b) )
// Block = 3 waves (gates i,g,o); wave w computes its gate tile for 4
// consecutive j-tiles (A-fragment reuse x4) over compile-time K; tiles staged
// in LDS, then the 96 threads apply the activation and store bf16 h directly.
// Grid = (H/64 = 16 j-groups, M/16 row tiles).
// ---------------------------------------------------------------------------
template <int K>
__global__ __launch_bounds__(96) void lstm1_fused_kernel(
    const bf16* __restrict__ A, const bf16* __restrict__ Wih,
    const float* __restrict__ bih, const float* __restrict__ bhh,
    bf16* __restrict__ hout)
{
  __shared__ float gt[3][4][16][16];
  const int wave = threadIdx.x >> 5;   // 0:i 1:g 2:o
  const int lane = threadIdx.x & 31;
  const int jg = blockIdx.x;           // j-tile group of 4 (0..15)
  const int mb = blockIdx.y;           // row tile (0..M/16-1)
  const int gatebase = (wave == 0) ? 0 : (wave == 1 ? 2048 : 3072);

  v8f acc[4];
#pragma unroll
  for (int j = 0; j < 4; ++j) {
    v8f z = {0.f, 0.f, 0.f, 0.f, 0.f, 0.f, 0.f, 0.f};
    acc[j] = z;
  }

  const bf16* Abase = A + (size_t)(mb * 16) * K;
  const bf16* Wbase = Wih + (size_t)(gatebase + jg * 64) * K;
#pragma unroll
  for (int k0 = 0; k0 < K; k0 += 32) {
    v16bf a = load_tile_frag(Abase + k0, K);
    v16bf b[4];
#pragma unroll
    for (int j = 0; j < 4; ++j)
      b[j] = load_tile_frag(Wbase + (size_t)j * 16 * K + k0, K);
#pragma unroll
    for (int j = 0; j < 4; ++j)
      acc[j] = __builtin_amdgcn_wmma_f32_16x16x32_bf16(
          false, a, false, b[j], (short)0, acc[j], false, false);
  }

  const int cn = lane & 15;
  const int rb = (lane >> 4) << 3;
#pragma unroll
  for (int j = 0; j < 4; ++j)
#pragma unroll
    for (int e = 0; e < 8; ++e) gt[wave][j][rb + e][cn] = acc[j][e];
  __syncthreads();

  for (int elem = threadIdx.x; elem < 1024; elem += 96) {
    int j  = elem >> 8;
    int r  = (elem >> 4) & 15;
    int cc = elem & 15;
    int col = jg * 64 + j * 16 + cc;     // 0..1023 within H
    int row = mb * 16 + r;
    float gi = gt[0][j][r][cc] + bih[col]        + bhh[col];
    float gg = gt[1][j][r][cc] + bih[2048 + col] + bhh[2048 + col];
    float go = gt[2][j][r][cc] + bih[3072 + col] + bhh[3072 + col];
    float c  = sigmoidf_(gi) * tanhf(gg);
    float h  = sigmoidf_(go) * tanhf(c);
    hout[(size_t)row * 1024 + col] = (bf16)h;
  }
}

// ---------------------------------------------------------------------------
// One recurrent LSTM timestep, fully fused:
//   gates(64x4096) = xg[:,t,:] + h_prev(64x1024) @ Whh(4096x1024)^T + biases
//   c = sig(f)*c + sig(i)*tanh(g);  h = sig(o)*tanh(c)
// Grid = (64 j-tiles, 4 b-tiles); block = 128 = 4 waves; wave w computes the
// 16x16 tile of gate w over K=1024 (32 WMMAs), stages it in LDS, then 128
// threads do the pointwise update.  h state double-buffered across launches.
// ---------------------------------------------------------------------------
__global__ __launch_bounds__(128) void lstm2_step_kernel(
    const bf16* __restrict__ hprev, bf16* __restrict__ hnext,
    float* __restrict__ cstate, const bf16* __restrict__ Whh,
    const float* __restrict__ xg, const float* __restrict__ bih,
    const float* __restrict__ bhh, bf16* __restrict__ hseq, int t)
{
  __shared__ float gtile[4][16][16];
  const int wave = threadIdx.x >> 5;
  const int lane = threadIdx.x & 31;
  const int nb = blockIdx.x;   // 0..63  (j tile in H)
  const int mb = blockIdx.y;   // 0..3   (batch tile)

  v8f acc = {0.f, 0.f, 0.f, 0.f, 0.f, 0.f, 0.f, 0.f};
  const bf16* Abase = hprev + (size_t)(mb * 16) * 1024;
  const bf16* Bbase = Whh + (size_t)(wave * 1024 + nb * 16) * 1024;
#pragma unroll
  for (int k0 = 0; k0 < 1024; k0 += 32) {
    v16bf a = load_tile_frag(Abase + k0, 1024);
    v16bf b = load_tile_frag(Bbase + k0, 1024);
    acc = __builtin_amdgcn_wmma_f32_16x16x32_bf16(
        false, a, false, b, (short)0, acc, false, false);
  }
  const int cn = lane & 15;
  const int rb = (lane >> 4) << 3;
#pragma unroll
  for (int e = 0; e < 8; ++e) gtile[wave][rb + e][cn] = acc[e];
  __syncthreads();

#pragma unroll
  for (int e = 0; e < 2; ++e) {
    int elem = threadIdx.x * 2 + e;          // 0..255
    int r = elem >> 4, cc = elem & 15;
    int b = mb * 16 + r;
    int j = nb * 16 + cc;
    size_t rm = (size_t)b * 256 + t;         // row in (B*T) token space
    const float* xr = xg + rm * 4096;
    float gi = gtile[0][r][cc] + xr[j]        + bih[j]        + bhh[j];
    float gf = gtile[1][r][cc] + xr[1024 + j] + bih[1024 + j] + bhh[1024 + j];
    float gg = gtile[2][r][cc] + xr[2048 + j] + bih[2048 + j] + bhh[2048 + j];
    float go = gtile[3][r][cc] + xr[3072 + j] + bih[3072 + j] + bhh[3072 + j];
    float c = sigmoidf_(gf) * cstate[b * 1024 + j] + sigmoidf_(gi) * tanhf(gg);
    float h = sigmoidf_(go) * tanhf(c);
    cstate[b * 1024 + j] = c;
    hnext[b * 1024 + j] = (bf16)h;
    if (hseq) hseq[rm * 1024 + j] = (bf16)h;
  }
}

// ---------------------------------------------------------------------------
// Final 64x512 @ 512x1 + bias  (tiny; plain VALU reduction)
// ---------------------------------------------------------------------------
__global__ __launch_bounds__(64) void mlp3_kernel(
    const bf16* __restrict__ z2, const bf16* __restrict__ w3,
    const float* __restrict__ b3, float* __restrict__ out)
{
  __shared__ float red[64];
  int b = blockIdx.x;
  float s = 0.f;
  for (int j = threadIdx.x; j < 512; j += 64)
    s += (float)z2[b * 512 + j] * (float)w3[j];
  red[threadIdx.x] = s;
  __syncthreads();
  for (int off = 32; off > 0; off >>= 1) {
    if (threadIdx.x < off) red[threadIdx.x] += red[threadIdx.x + off];
    __syncthreads();
  }
  if (threadIdx.x == 0) out[b] = red[0] + b3[0];
}

// ---------------------------------------------------------------------------
// Host orchestration
// ---------------------------------------------------------------------------
extern "C" void kernel_launch(void* const* d_in, const int* in_sizes, int n_in,
                              void* d_out, int out_size, void* d_ws, size_t ws_size,
                              hipStream_t stream)
{
  (void)in_sizes; (void)n_in; (void)out_size; (void)ws_size;
  constexpr int B = 64, T = 256, D = 256, H = 1024;
  constexpr int M  = B * T;       // 16384 token rows
  constexpr int G4 = 4 * H;       // 4096 gate columns

  const float* xx       = (const float*)d_in[0];
  const float* l1_Wih0  = (const float*)d_in[1];
  const float* l1_bih0  = (const float*)d_in[2];
  const float* l1_bhh0  = (const float*)d_in[3];
  const float* l1_Wih1  = (const float*)d_in[4];
  const float* l1_bih1  = (const float*)d_in[5];
  const float* l1_bhh1  = (const float*)d_in[6];
  const float* l2_Wih0  = (const float*)d_in[7];
  const float* l2_Whh0  = (const float*)d_in[8];
  const float* l2_bih0  = (const float*)d_in[9];
  const float* l2_bhh0  = (const float*)d_in[10];
  const float* l2_Wih1  = (const float*)d_in[11];
  const float* l2_Whh1  = (const float*)d_in[12];
  const float* l2_bih1  = (const float*)d_in[13];
  const float* l2_bhh1  = (const float*)d_in[14];
  const float* mlp_W1   = (const float*)d_in[15];
  const float* mlp_b1   = (const float*)d_in[16];
  const float* mlp_W2   = (const float*)d_in[17];
  const float* mlp_b2   = (const float*)d_in[18];
  const float* mlp_W3   = (const float*)d_in[19];
  const float* mlp_b3   = (const float*)d_in[20];
  float* out = (float*)d_out;

  char* ws = (char*)d_ws;
  size_t off = 0;
  auto alloc = [&](size_t bytes) -> char* {
    char* p = ws + off;
    off += (bytes + 255) & ~(size_t)255;
    return p;
  };

  bf16* xbf      = (bf16*)alloc((size_t)M * D * 2);
  bf16* wl1a     = (bf16*)alloc((size_t)G4 * D * 2);
  bf16* wl1b     = (bf16*)alloc((size_t)G4 * H * 2);
  bf16* wl2ihA   = (bf16*)alloc((size_t)G4 * H * 2);
  bf16* wl2hhA   = (bf16*)alloc((size_t)G4 * H * 2);
  bf16* wl2ihB   = (bf16*)alloc((size_t)G4 * H * 2);
  bf16* wl2hhB   = (bf16*)alloc((size_t)G4 * H * 2);
  bf16* wm1      = (bf16*)alloc((size_t)H * H * 2);
  bf16* wm2      = (bf16*)alloc((size_t)(H / 2) * H * 2);
  bf16* wm3      = (bf16*)alloc((size_t)(H / 2) * 2);
  bf16* hseqA    = (bf16*)alloc((size_t)M * H * 2);
  bf16* hseqB    = (bf16*)alloc((size_t)M * H * 2);
  float* xg      = (float*)alloc((size_t)M * G4 * 4);
  bf16* hstate   = (bf16*)alloc((size_t)2 * B * H * 2);   // double-buffered
  float* cstate  = (float*)alloc((size_t)B * H * 4);
  bf16* z1       = (bf16*)alloc((size_t)B * H * 2);
  bf16* z2       = (bf16*)alloc((size_t)B * (H / 2) * 2);

  auto cvt = [&](const float* src, bf16* dst, int n) {
    f2bf_kernel<<<(n + 255) / 256, 256, 0, stream>>>(src, dst, n);
  };
  auto gemm_f32 = [&](const bf16* A, const bf16* W, float* o, int m, int n) {
    gemm_bf16_wmma<0, H><<<dim3(n / 512, m / 32), 256, 0, stream>>>(
        A, W, o, nullptr, nullptr, m, n);
  };
  auto gemm_relu = [&](const bf16* A, const bf16* W, bf16* o, const float* bias,
                       int m, int n) {
    gemm_bf16_wmma<1, H><<<dim3(n / 512, m / 32), 256, 0, stream>>>(
        A, W, nullptr, o, bias, m, n);
  };

  // --- weight / input conversion to bf16 -----------------------------------
  cvt(xx, xbf, M * D);
  cvt(l1_Wih0, wl1a, G4 * D);
  cvt(l1_Wih1, wl1b, G4 * H);
  cvt(l2_Wih0, wl2ihA, G4 * H);
  cvt(l2_Whh0, wl2hhA, G4 * H);
  cvt(l2_Wih1, wl2ihB, G4 * H);
  cvt(l2_Whh1, wl2hhB, G4 * H);
  cvt(mlp_W1, wm1, H * H);
  cvt(mlp_W2, wm2, (H / 2) * H);
  cvt(mlp_W3, wm3, H / 2);

  // --- layers 1 & 2 (no recurrence): fully fused, f-gate skipped -----------
  lstm1_fused_kernel<D><<<dim3(H / 64, M / 16), 96, 0, stream>>>(
      xbf, wl1a, l1_bih0, l1_bhh0, hseqA);
  lstm1_fused_kernel<H><<<dim3(H / 64, M / 16), 96, 0, stream>>>(
      hseqA, wl1b, l1_bih1, l1_bhh1, hseqB);

  // --- recurrent layer A: input projection, then 256 fused steps -----------
  gemm_f32(hseqB, wl2ihA, xg, M, G4);
  hipMemsetAsync(hstate, 0, (size_t)2 * B * H * 2, stream);
  hipMemsetAsync(cstate, 0, (size_t)B * H * 4, stream);
  for (int t = 0; t < T; ++t) {
    lstm2_step_kernel<<<dim3(H / 16, B / 16), 128, 0, stream>>>(
        hstate + (size_t)(t & 1) * B * H, hstate + (size_t)((t + 1) & 1) * B * H,
        cstate, wl2hhA, xg, l2_bih0, l2_bhh0, hseqA, t);
  }

  // --- recurrent layer B ----------------------------------------------------
  gemm_f32(hseqA, wl2ihB, xg, M, G4);
  hipMemsetAsync(hstate, 0, (size_t)2 * B * H * 2, stream);
  hipMemsetAsync(cstate, 0, (size_t)B * H * 4, stream);
  for (int t = 0; t < T; ++t) {
    lstm2_step_kernel<<<dim3(H / 16, B / 16), 128, 0, stream>>>(
        hstate + (size_t)(t & 1) * B * H, hstate + (size_t)((t + 1) & 1) * B * H,
        cstate, wl2hhB, xg, l2_bih1, l2_bhh1, nullptr, t);
  }
  // final h (t=255 writes buffer index (255+1)&1 == 0)
  const bf16* hlast = hstate;

  // --- MLP head (WMMA + fused bias/ReLU) -----------------------------------
  gemm_relu(hlast, wm1, z1, mlp_b1, B, H);             // 64x1024
  gemm_relu(z1, wm2, z2, mlp_b2, B, H / 2);            // 64x512
  mlp3_kernel<<<B, 64, 0, stream>>>(z2, wm3, mlp_b3, out);
}
